// SpatialTransformer_3942779977830
// MI455X (gfx1250) — matrix operand order repaired and checked by reading, not compiled, not run on hardware
//
#include <hip/hip_runtime.h>

typedef unsigned short bf16;
typedef __attribute__((ext_vector_type(16))) __bf16 v16bf;
typedef __attribute__((ext_vector_type(8)))  float  v8f;

// ---------- bf16 helpers (round-to-nearest-even) ----------
__device__ __forceinline__ bf16 f2b(float f) {
  union { float f; unsigned u; } c; c.f = f;
  unsigned u = c.u;
  u += 0x7fffu + ((u >> 16) & 1u);
  return (bf16)(u >> 16);
}
__device__ __forceinline__ float b2f(bf16 h) {
  union { unsigned u; float f; } c; c.u = ((unsigned)h) << 16;
  return c.f;
}

// =====================================================================
// Weight transpose + convert: src f32 [K,N] -> dst bf16 [N,K]
// =====================================================================
__global__ __launch_bounds__(256) void k_wtrans(const float* __restrict__ src,
                                                bf16* __restrict__ dst,
                                                int K, int N) {
  size_t idx = (size_t)blockIdx.x * 256 + threadIdx.x;
  size_t tot = (size_t)K * N;
  if (idx >= tot) return;
  int k = (int)(idx / N);
  int n = (int)(idx % N);
  dst[(size_t)n * K + k] = f2b(src[idx]);
}

// f32 -> bf16 flat convert
__global__ __launch_bounds__(256) void k_f2b(const float* __restrict__ src,
                                             bf16* __restrict__ dst, size_t n) {
  size_t idx = (size_t)blockIdx.x * 256 + threadIdx.x;
  if (idx < n) dst[idx] = f2b(src[idx]);
}

// =====================================================================
// GroupNorm: x f32 [B=8,C=640,HW=1024] -> bf16 [B, HW, C] (token major)
// one block per (b, group); 20 channels * 1024 pixels per group
// =====================================================================
__global__ __launch_bounds__(256) void k_groupnorm(const float* __restrict__ x,
                                                   const float* __restrict__ gamma,
                                                   const float* __restrict__ beta,
                                                   bf16* __restrict__ out) {
  const int b = blockIdx.x >> 5;
  const int g = blockIdx.x & 31;
  const float* xb = x + ((size_t)b * 640 + g * 20) * 1024;
  float s = 0.f, ss = 0.f;
  for (int i = threadIdx.x; i < 20480; i += 256) {
    float v = xb[i]; s += v; ss += v * v;
  }
  __shared__ float sh[256], sh2[256];
  sh[threadIdx.x] = s; sh2[threadIdx.x] = ss;
  __syncthreads();
  for (int o = 128; o > 0; o >>= 1) {
    if (threadIdx.x < o) { sh[threadIdx.x] += sh[threadIdx.x + o]; sh2[threadIdx.x] += sh2[threadIdx.x + o]; }
    __syncthreads();
  }
  float mu = sh[0] * (1.0f / 20480.0f);
  float var = sh2[0] * (1.0f / 20480.0f) - mu * mu;
  float rstd = rsqrtf(var + 1e-6f);
  for (int i = threadIdx.x; i < 20480; i += 256) {
    int c = g * 20 + (i >> 10);
    int sp = i & 1023;
    float v = (xb[i] - mu) * rstd * gamma[c] + beta[c];
    out[((size_t)b * 1024 + sp) * 640 + c] = f2b(v);
  }
}

// =====================================================================
// LayerNorm: t f32 [M,640] -> bf16 [M,640]; one block per row
// =====================================================================
__global__ __launch_bounds__(256) void k_layernorm(const float* __restrict__ t,
                                                   const float* __restrict__ g,
                                                   const float* __restrict__ b,
                                                   bf16* __restrict__ out) {
  const size_t row = blockIdx.x;
  const float* tr = t + row * 640;
  float s = 0.f, ss = 0.f;
  for (int i = threadIdx.x; i < 640; i += 256) { float v = tr[i]; s += v; ss += v * v; }
  __shared__ float sh[256], sh2[256];
  sh[threadIdx.x] = s; sh2[threadIdx.x] = ss;
  __syncthreads();
  for (int o = 128; o > 0; o >>= 1) {
    if (threadIdx.x < o) { sh[threadIdx.x] += sh[threadIdx.x + o]; sh2[threadIdx.x] += sh2[threadIdx.x + o]; }
    __syncthreads();
  }
  float mu = sh[0] * (1.0f / 640.0f);
  float var = sh2[0] * (1.0f / 640.0f) - mu * mu;
  float rstd = rsqrtf(var + 1e-5f);
  for (int i = threadIdx.x; i < 640; i += 256)
    out[row * 640 + i] = f2b((tr[i] - mu) * rstd * g[i] + b[i]);
}

// =====================================================================
// GEGLU: proj bf16 [rows,5120] -> out bf16 [rows,2560], a * gelu(gate)
// =====================================================================
__global__ __launch_bounds__(256) void k_geglu(const bf16* __restrict__ proj,
                                               bf16* __restrict__ out, size_t rows) {
  size_t idx = (size_t)blockIdx.x * 256 + threadIdx.x;
  size_t tot = rows * 2560;
  if (idx >= tot) return;
  size_t row = idx / 2560;
  int c = (int)(idx % 2560);
  float a = b2f(proj[row * 5120 + c]);
  float gt = b2f(proj[row * 5120 + 2560 + c]);
  float ge = 0.5f * gt * (1.0f + erff(gt * 0.70710678118654752f));
  out[idx] = f2b(a * ge);
}

// =====================================================================
// WMMA GEMM:  D[M,N] = A[M,K](bf16) * Bt[N,K](bf16)^T  (+bias, +resid)
// block = 256 threads = 8 waves; tile 128(M) x 64(N); K step 32.
// B tile staged in LDS, shared by all 8 waves.
// epilogue modes:
//   0: out[row*ldc+col] (+resid) -> outF and/or outB
//   1: head-padded Q/K store: [(b*8+h)*p0 + n]*96 + d  (b=row/p1, n=row%p1)
//   2: V-transposed store:    [(b*8+h)*96 + d]*p0 + n
//   3: final: outF[(b*640+col)*1024 + n] = v + resid[idx]  (resid = x_in)
// =====================================================================
__global__ __launch_bounds__(256) void k_gemm(
    const bf16* __restrict__ A, int lda,
    const bf16* __restrict__ Bt, int ldb,
    const float* __restrict__ bias,
    const float* __restrict__ resid,
    float* __restrict__ outF, bf16* __restrict__ outB,
    int M, int N, int K, int ldc, int mode, int p0, int p1) {
  __shared__ __align__(16) bf16 Bs[64][32];
  const int tid = threadIdx.x;
  const int wid = tid >> 5;
  const int l = tid & 31;
  const int m0 = blockIdx.y * 128 + wid * 16;
  const int n0 = blockIdx.x * 64;
  const int mm = l & 15;
  const int kbA = (l < 16) ? 0 : 16;
  const int kbB = (l < 16) ? 0 : 16;

  v8f acc[4];
#pragma unroll
  for (int t = 0; t < 4; ++t) acc[t] = (v8f){0,0,0,0,0,0,0,0};

  int ar = m0 + mm; if (ar >= M) ar = M - 1;
  const bf16* aRow = A + (size_t)ar * lda;

  for (int k0 = 0; k0 < K; k0 += 32) {
    __syncthreads();
    {
      int row = tid >> 2, ch = tid & 3;
      *(uint4*)&Bs[row][ch * 8] =
          *(const uint4*)&Bt[(size_t)(n0 + row) * ldb + k0 + ch * 8];
    }
    __syncthreads();
    union { v16bf v; uint4 u[2]; } fa;
    const bf16* ap = aRow + k0 + kbA;
    fa.u[0] = *(const uint4*)(ap);
    fa.u[1] = *(const uint4*)(ap + 8);
#pragma unroll
    for (int t = 0; t < 4; ++t) {
      union { v16bf v; uint4 u[2]; } fb;
      const bf16* bp = &Bs[t * 16 + mm][kbB];
      fb.u[0] = *(const uint4*)(bp);
      fb.u[1] = *(const uint4*)(bp + 8);
      acc[t] = __builtin_amdgcn_wmma_f32_16x16x32_bf16(
          false, fa.v, false, fb.v, (short)0, acc[t], false, false);
    }
  }

  const int rbase = (l < 16) ? 0 : 8;
#pragma unroll
  for (int t = 0; t < 4; ++t) {
    int col = n0 + t * 16 + mm;
    float bv = bias ? bias[col] : 0.0f;
#pragma unroll
    for (int r = 0; r < 8; ++r) {
      int row = m0 + rbase + r;
      if (row >= M) continue;
      float v = acc[t][r] + bv;
      if (mode == 0) {
        size_t idx = (size_t)row * ldc + col;
        if (resid) v += resid[idx];
        if (outF) outF[idx] = v;
        if (outB) outB[idx] = f2b(v);
      } else if (mode == 1) {
        int b = row / p1, n = row % p1;
        int h = col / 80, d = col % 80;
        outB[(((size_t)b * 8 + h) * p0 + n) * 96 + d] = f2b(v);
      } else if (mode == 2) {
        int b = row / p1, n = row % p1;
        int h = col / 80, d = col % 80;
        outB[(((size_t)b * 8 + h) * 96 + d) * (size_t)p0 + n] = f2b(v);
      } else {
        int b = row >> 10, n = row & 1023;
        size_t idx = ((size_t)b * 640 + col) * 1024 + n;
        outF[idx] = v + resid[idx];
      }
    }
  }
}

// =====================================================================
// Flash-style attention (WMMA sim + WMMA AV, online softmax).
// Q  : [B*8, 1024, 96] bf16 (dhead 80 zero-padded to 96)
// Kp : [B*8, Mpad, 96] bf16 (zero-padded)
// Vt : [B*8, 96, Mpad]  bf16 (d-major, tokens contiguous, zero-padded)
// Out: [B, 1024, 640]   bf16
// block = 128 threads (4 waves), each wave owns 16 query rows.
// =====================================================================
__global__ __launch_bounds__(128) void k_attn(
    const bf16* __restrict__ Q, const bf16* __restrict__ Kp,
    const bf16* __restrict__ Vt, bf16* __restrict__ Out,
    int Mkv, int Mpad, float scale) {
  __shared__ __align__(16) bf16 P[4][16][32];
  const int bh = blockIdx.y;
  const int b = bh >> 3, h = bh & 7;
  const int wid = threadIdx.x >> 5;
  const int l = threadIdx.x & 31;
  const int q0 = blockIdx.x * 64 + wid * 16;
  const int mm = l & 15;
  const int kbA = (l < 16) ? 0 : 16;
  const int kbB = (l < 16) ? 0 : 16;
  const int rbase = (l < 16) ? 0 : 8;

  // Q fragments for 3 K-steps (K = 96)
  const bf16* qbase = Q + ((size_t)bh * 1024 + q0) * 96 + (size_t)mm * 96;
  union { v16bf v; uint4 u[2]; } qa[3];
#pragma unroll
  for (int kk = 0; kk < 3; ++kk) {
    const bf16* p = qbase + kk * 32 + kbA;
    qa[kk].u[0] = *(const uint4*)p;
    qa[kk].u[1] = *(const uint4*)(p + 8);
  }

  float rm[8], rs[8];
#pragma unroll
  for (int r = 0; r < 8; ++r) { rm[r] = -1e30f; rs[r] = 0.0f; }
  v8f o[5];
#pragma unroll
  for (int t = 0; t < 5; ++t) o[t] = (v8f){0,0,0,0,0,0,0,0};

  for (int j0 = 0; j0 < Mpad; j0 += 32) {
    v8f s0 = (v8f){0,0,0,0,0,0,0,0};
    v8f s1 = (v8f){0,0,0,0,0,0,0,0};
    const bf16* kb0 = Kp + ((size_t)bh * Mpad + j0 + mm) * 96;
    const bf16* kb1 = kb0 + 16 * 96;
#pragma unroll
    for (int kk = 0; kk < 3; ++kk) {
      union { v16bf v; uint4 u[2]; } fb;
      const bf16* p = kb0 + kk * 32 + kbB;
      fb.u[0] = *(const uint4*)p;
      fb.u[1] = *(const uint4*)(p + 8);
      s0 = __builtin_amdgcn_wmma_f32_16x16x32_bf16(false, qa[kk].v, false, fb.v,
                                                   (short)0, s0, false, false);
    }
#pragma unroll
    for (int kk = 0; kk < 3; ++kk) {
      union { v16bf v; uint4 u[2]; } fb;
      const bf16* p = kb1 + kk * 32 + kbB;
      fb.u[0] = *(const uint4*)p;
      fb.u[1] = *(const uint4*)(p + 8);
      s1 = __builtin_amdgcn_wmma_f32_16x16x32_bf16(false, qa[kk].v, false, fb.v,
                                                   (short)0, s1, false, false);
    }
    const bool mask0 = (j0 + mm) >= Mkv;
    const bool mask1 = (j0 + 16 + mm) >= Mkv;
#pragma unroll
    for (int r = 0; r < 8; ++r) {
      float x0 = mask0 ? -1e30f : s0[r] * scale;
      float x1 = mask1 ? -1e30f : s1[r] * scale;
      float m = fmaxf(x0, x1);
      m = fmaxf(m, __shfl_xor(m, 1, 16));
      m = fmaxf(m, __shfl_xor(m, 2, 16));
      m = fmaxf(m, __shfl_xor(m, 4, 16));
      m = fmaxf(m, __shfl_xor(m, 8, 16));
      float mn = fmaxf(rm[r], m);
      float corr = __expf(rm[r] - mn);
      float p0v = __expf(x0 - mn);
      float p1v = __expf(x1 - mn);
      float sr = p0v + p1v;
      sr += __shfl_xor(sr, 1, 16);
      sr += __shfl_xor(sr, 2, 16);
      sr += __shfl_xor(sr, 4, 16);
      sr += __shfl_xor(sr, 8, 16);
      rs[r] = rs[r] * corr + sr;
      rm[r] = mn;
#pragma unroll
      for (int dt = 0; dt < 5; ++dt) o[dt][r] *= corr;
      P[wid][rbase + r][mm] = f2b(p0v);
      P[wid][rbase + r][16 + mm] = f2b(p1v);
    }
    __syncthreads();
    union { v16bf v; uint4 u[2]; } pa;
    {
      const bf16* pp = &P[wid][mm][kbA];
      pa.u[0] = *(const uint4*)pp;
      pa.u[1] = *(const uint4*)(pp + 8);
    }
#pragma unroll
    for (int dt = 0; dt < 5; ++dt) {
      union { v16bf v; uint4 u[2]; } vb;
      const bf16* vp = Vt + ((size_t)bh * 96 + dt * 16 + mm) * Mpad + j0 + kbB;
      vb.u[0] = *(const uint4*)vp;
      vb.u[1] = *(const uint4*)(vp + 8);
      o[dt] = __builtin_amdgcn_wmma_f32_16x16x32_bf16(false, pa.v, false, vb.v,
                                                      (short)0, o[dt], false, false);
    }
    __syncthreads();
  }
#pragma unroll
  for (int r = 0; r < 8; ++r) {
    float inv = 1.0f / rs[r];
    size_t obase = ((size_t)b * 1024 + q0 + rbase + r) * 640 + h * 80;
#pragma unroll
    for (int dt = 0; dt < 5; ++dt)
      Out[obase + dt * 16 + mm] = f2b(o[dt][r] * inv);
  }
}

// =====================================================================
// Host launcher
// =====================================================================
extern "C" void kernel_launch(void* const* d_in, const int* in_sizes, int n_in,
                              void* d_out, int out_size, void* d_ws, size_t ws_size,
                              hipStream_t stream) {
  (void)in_sizes; (void)n_in; (void)out_size; (void)ws_size;
  const float* x       = (const float*)d_in[0];
  const float* ctx     = (const float*)d_in[1];
  const float* gn_g    = (const float*)d_in[2];
  const float* gn_b    = (const float*)d_in[3];
  const float* w_in    = (const float*)d_in[4];
  const float* b_in    = (const float*)d_in[5];
  const float* ln1_g   = (const float*)d_in[6];
  const float* ln1_b   = (const float*)d_in[7];
  const float* wq1     = (const float*)d_in[8];
  const float* wk1     = (const float*)d_in[9];
  const float* wv1     = (const float*)d_in[10];
  const float* wo1     = (const float*)d_in[11];
  const float* bo1     = (const float*)d_in[12];
  const float* ln2_g   = (const float*)d_in[13];
  const float* ln2_b   = (const float*)d_in[14];
  const float* wq2     = (const float*)d_in[15];
  const float* wk2     = (const float*)d_in[16];
  const float* wv2     = (const float*)d_in[17];
  const float* wo2     = (const float*)d_in[18];
  const float* bo2     = (const float*)d_in[19];
  const float* ln3_g   = (const float*)d_in[20];
  const float* ln3_b   = (const float*)d_in[21];
  const float* w_geglu = (const float*)d_in[22];
  const float* b_geglu = (const float*)d_in[23];
  const float* w_ffo   = (const float*)d_in[24];
  const float* b_ffo   = (const float*)d_in[25];
  const float* w_out   = (const float*)d_in[26];
  const float* b_out   = (const float*)d_in[27];
  float* out = (float*)d_out;

  const int MROWS = 8 * 1024;          // 8192 tokens
  const int MCTX  = 8 * 77;            // 616 context rows
  char* wsp = (char*)d_ws;
  auto alloc = [&](size_t bytes) {
    char* p = wsp;
    wsp += (bytes + 255) & ~(size_t)255;
    return p;
  };

  bf16* wt_in    = (bf16*)alloc((size_t)640 * 640 * 2);
  bf16* wt_q1    = (bf16*)alloc((size_t)640 * 640 * 2);
  bf16* wt_k1    = (bf16*)alloc((size_t)640 * 640 * 2);
  bf16* wt_v1    = (bf16*)alloc((size_t)640 * 640 * 2);
  bf16* wt_o1    = (bf16*)alloc((size_t)640 * 640 * 2);
  bf16* wt_q2    = (bf16*)alloc((size_t)640 * 640 * 2);
  bf16* wt_k2    = (bf16*)alloc((size_t)640 * 768 * 2);
  bf16* wt_v2    = (bf16*)alloc((size_t)640 * 768 * 2);
  bf16* wt_o2    = (bf16*)alloc((size_t)640 * 640 * 2);
  bf16* wt_geglu = (bf16*)alloc((size_t)5120 * 640 * 2);
  bf16* wt_ffo   = (bf16*)alloc((size_t)640 * 2560 * 2);
  bf16* wt_out   = (bf16*)alloc((size_t)640 * 640 * 2);
  bf16* act_b    = (bf16*)alloc((size_t)MROWS * 640 * 2);   // gn-out / tn
  bf16* ctx_b    = (bf16*)alloc((size_t)MCTX * 768 * 2);
  float* t_f     = (float*)alloc((size_t)MROWS * 640 * 4);  // residual stream
  const size_t QP_BYTES = (size_t)8 * 8 * 1024 * 96 * 2;
  bf16* qp       = (bf16*)alloc(QP_BYTES);
  bf16* kp       = (bf16*)alloc(QP_BYTES);
  bf16* vt       = (bf16*)alloc(QP_BYTES);
  bf16* attn_b   = (bf16*)alloc((size_t)MROWS * 640 * 2);
  bf16* proj_b   = (bf16*)alloc((size_t)MROWS * 5120 * 2);
  bf16* gg_b     = (bf16*)alloc((size_t)MROWS * 2560 * 2);

  auto WT = [&](const float* s, bf16* d, int K, int N) {
    size_t tot = (size_t)K * N;
    k_wtrans<<<dim3((unsigned)((tot + 255) / 256)), dim3(256), 0, stream>>>(s, d, K, N);
  };
  auto GEMM = [&](const bf16* A, int lda, const bf16* Bt, int ldb,
                  const float* bias, const float* resid, float* of, bf16* ob,
                  int M, int N, int K, int ldc, int mode, int p0, int p1) {
    dim3 g(N / 64, (M + 127) / 128);
    k_gemm<<<g, dim3(256), 0, stream>>>(A, lda, Bt, ldb, bias, resid, of, ob,
                                        M, N, K, ldc, mode, p0, p1);
  };

  // ---- weight prep (bf16, transposed to [N,K]) ----
  WT(w_in, wt_in, 640, 640);
  WT(wq1, wt_q1, 640, 640);
  WT(wk1, wt_k1, 640, 640);
  WT(wv1, wt_v1, 640, 640);
  WT(wo1, wt_o1, 640, 640);
  WT(wq2, wt_q2, 640, 640);
  WT(wk2, wt_k2, 768, 640);
  WT(wv2, wt_v2, 768, 640);
  WT(wo2, wt_o2, 640, 640);
  WT(w_geglu, wt_geglu, 640, 5120);
  WT(w_ffo, wt_ffo, 2560, 640);
  WT(w_out, wt_out, 640, 640);
  {
    size_t n = (size_t)MCTX * 768;
    k_f2b<<<dim3((unsigned)((n + 255) / 256)), dim3(256), 0, stream>>>(ctx, ctx_b, n);
  }

  // ---- GroupNorm + implicit [b,c,hw]->[b,hw,c] transpose ----
  k_groupnorm<<<dim3(8 * 32), dim3(256), 0, stream>>>(x, gn_g, gn_b, act_b);

  hipMemsetAsync(qp, 0, QP_BYTES, stream);
  hipMemsetAsync(kp, 0, QP_BYTES, stream);
  hipMemsetAsync(vt, 0, QP_BYTES, stream);

  // ---- proj_in ----
  GEMM(act_b, 640, wt_in, 640, b_in, nullptr, t_f, nullptr,
       MROWS, 640, 640, 640, 0, 0, 0);

  // ---- self attention ----
  k_layernorm<<<dim3(MROWS), dim3(256), 0, stream>>>(t_f, ln1_g, ln1_b, act_b);
  GEMM(act_b, 640, wt_q1, 640, nullptr, nullptr, nullptr, qp, MROWS, 640, 640, 0, 1, 1024, 1024);
  GEMM(act_b, 640, wt_k1, 640, nullptr, nullptr, nullptr, kp, MROWS, 640, 640, 0, 1, 1024, 1024);
  GEMM(act_b, 640, wt_v1, 640, nullptr, nullptr, nullptr, vt, MROWS, 640, 640, 0, 2, 1024, 1024);
  k_attn<<<dim3(16, 64), dim3(128), 0, stream>>>(qp, kp, vt, attn_b, 1024, 1024,
                                                 0.11180339887498949f);
  GEMM(attn_b, 640, wt_o1, 640, bo1, t_f, t_f, nullptr, MROWS, 640, 640, 640, 0, 0, 0);

  // ---- cross attention ----
  k_layernorm<<<dim3(MROWS), dim3(256), 0, stream>>>(t_f, ln2_g, ln2_b, act_b);
  hipMemsetAsync(kp, 0, QP_BYTES, stream);
  hipMemsetAsync(vt, 0, QP_BYTES, stream);
  GEMM(act_b, 640, wt_q2, 640, nullptr, nullptr, nullptr, qp, MROWS, 640, 640, 0, 1, 1024, 1024);
  GEMM(ctx_b, 768, wt_k2, 768, nullptr, nullptr, nullptr, kp, MCTX, 640, 768, 0, 1, 96, 77);
  GEMM(ctx_b, 768, wt_v2, 768, nullptr, nullptr, nullptr, vt, MCTX, 640, 768, 0, 2, 96, 77);
  k_attn<<<dim3(16, 64), dim3(128), 0, stream>>>(qp, kp, vt, attn_b, 77, 96,
                                                 0.11180339887498949f);
  GEMM(attn_b, 640, wt_o2, 640, bo2, t_f, t_f, nullptr, MROWS, 640, 640, 640, 0, 0, 0);

  // ---- GEGLU feed-forward ----
  k_layernorm<<<dim3(MROWS), dim3(256), 0, stream>>>(t_f, ln3_g, ln3_b, act_b);
  GEMM(act_b, 640, wt_geglu, 640, b_geglu, nullptr, nullptr, proj_b,
       MROWS, 5120, 640, 5120, 0, 0, 0);
  {
    size_t tot = (size_t)MROWS * 2560;
    k_geglu<<<dim3((unsigned)((tot + 255) / 256)), dim3(256), 0, stream>>>(proj_b, gg_b, MROWS);
  }
  GEMM(gg_b, 2560, wt_ffo, 2560, b_ffo, t_f, t_f, nullptr, MROWS, 640, 2560, 640, 0, 0, 0);

  // ---- proj_out (+ transpose back + input residual) ----
  {
    size_t n = (size_t)MROWS * 640;
    k_f2b<<<dim3((unsigned)((n + 255) / 256)), dim3(256), 0, stream>>>(t_f, act_b, n);
  }
  GEMM(act_b, 640, wt_out, 640, b_out, x, out, nullptr, MROWS, 640, 640, 0, 3, 0, 0);
}
